// NNFMLoss_63565515981341
// MI455X (gfx1250) — matrix operand order repaired and checked by reading, not compiled, not run on hardware
//
#include <hip/hip_runtime.h>
#include <hip/hip_bf16.h>

#define HW   16384
#define CCH  768
#define EPSF 1e-8f

typedef __attribute__((ext_vector_type(16))) __bf16 v16bf;
typedef __attribute__((ext_vector_type(8)))  float  v8f;

static __device__ __forceinline__ unsigned short f2bf(float f) {
    unsigned int u = __float_as_uint(f);
    u += 0x7FFFu + ((u >> 16) & 1u);     // round-to-nearest-even
    return (unsigned short)(u >> 16);
}

// CDNA5 async global->LDS copy (ASYNCcnt-tracked), 16B per lane.
static __device__ __forceinline__ void async_copy_b128(unsigned lds_byte_off,
                                                       const void* gaddr) {
    asm volatile("global_load_async_to_lds_b128 %0, %1, off"
                 :: "v"(lds_byte_off), "v"(gaddr) : "memory");
}
static __device__ __forceinline__ void wait_async0() {
    asm volatile("s_wait_asynccnt 0" ::: "memory");
}

// ---------------------------------------------------------------------------
// 1) Per-position channel norms for a and b (coalesced over positions).
// ---------------------------------------------------------------------------
__global__ void norms_kernel(const float* __restrict__ a, const float* __restrict__ b,
                             float* __restrict__ normA, float* __restrict__ denomA,
                             float* __restrict__ normB, float* __restrict__ denomB) {
    int p = blockIdx.x * blockDim.x + threadIdx.x;
    float sa = 0.f, sb = 0.f;
    for (int c = 0; c < CCH; ++c) {
        float va = a[(size_t)c * HW + p];
        float vb = b[(size_t)c * HW + p];
        sa = fmaf(va, va, sa);
        sb = fmaf(vb, vb, sb);
    }
    normA[p]  = sqrtf(sa);
    denomA[p] = sqrtf(sa + EPSF) + EPSF;
    normB[p]  = sqrtf(sb);
    denomB[p] = sqrtf(sb + EPSF) + EPSF;
}

// ---------------------------------------------------------------------------
// 2) Normalize + transpose [c][p] f32 -> [p][c] bf16 via LDS tile.
// ---------------------------------------------------------------------------
__global__ void xpose_kernel(const float* __restrict__ src, const float* __restrict__ denom,
                             unsigned short* __restrict__ dst) {
    __shared__ float tile[32 * 33];
    int tx = threadIdx.x & 31, ty = threadIdx.x >> 5;
    int p0 = blockIdx.x * 32, c0 = blockIdx.y * 32;
#pragma unroll
    for (int cc = 0; cc < 4; ++cc) {
        int cl = ty + cc * 8;
        tile[cl * 33 + tx] = src[(size_t)(c0 + cl) * HW + (p0 + tx)];
    }
    __syncthreads();
#pragma unroll
    for (int pc = 0; pc < 4; ++pc) {
        int pl = ty + pc * 8;
        float inv = 1.0f / denom[p0 + pl];
        dst[(size_t)(p0 + pl) * CCH + (c0 + tx)] = f2bf(tile[tx * 33 + pl] * inv);
    }
}

// ---------------------------------------------------------------------------
// 3) Argmax-GEMM: S = An * Bn^T (both stored [pos][ch] bf16), row-wise argmax.
//    Block: 256 thr = 8 waves in 4(M)x2(N); wave tile 32x64; block tile 128x128.
//    B tile staged in LDS by async-copy (double buffered, K-chunks of 64).
// ---------------------------------------------------------------------------
#define KCHUNK      64
#define NCHUNK      (CCH / KCHUNK)          // 12
#define BROW_STRIDE 72                      // halfs; padded vs 64 to dodge bank conflicts
#define CHUNK_HALFS (128 * BROW_STRIDE)     // 9216 halfs = 18 KB per buffer

__launch_bounds__(256)
__global__ void argmax_gemm_kernel(const unsigned short* __restrict__ An,
                                   const unsigned short* __restrict__ Bn,
                                   float* __restrict__ pmax, int* __restrict__ pidx) {
    // 36 KB: double-buffered B staging; aliased by the final argmax reduction.
    __shared__ __align__(16) char smem[2 * CHUNK_HALFS * 2];
    unsigned short* bsmem = (unsigned short*)smem;
    const unsigned  lds_base = (unsigned)(uintptr_t)(void*)smem;

    const int tid  = threadIdx.x;
    const int lane = tid & 31;
    const int wid  = tid >> 5;
    const int wm   = wid & 3;        // M-group (32 rows)
    const int wn   = wid >> 2;       // N-group (64 cols)
    const int l15  = lane & 15;
    const int hsel = lane >> 4;      // half-wave select

    const int iBase = blockIdx.x * 128 + wm * 32;
    const int jt0   = blockIdx.y * 4096;

    // Cooperative-copy mapping: thread -> (row, 64B segment) of the 128x64 chunk.
    const int crow = tid >> 1;            // 0..127
    const int cseg = (tid & 1) * 32;      // half offset within row (halfs)

    // A fragment row pointers (16-bit A 16x32 layout: lane<16 K0..7,16..23; lane>=16 K8..15,24..31)
    const unsigned short* aptr[2];
#pragma unroll
    for (int mi = 0; mi < 2; ++mi)
        aptr[mi] = An + (size_t)(iBase + mi * 16 + l15) * CCH;

    float rmax[2][8];
    int   ridx[2][8];
#pragma unroll
    for (int mi = 0; mi < 2; ++mi)
#pragma unroll
        for (int r = 0; r < 8; ++r) { rmax[mi][r] = -3.4e38f; ridx[mi][r] = 0; }

    for (int jt = jt0; jt < jt0 + 4096; jt += 128) {
        // ---- prefetch chunk 0 into buffer 0 ----
        {
            const unsigned short* src = Bn + (size_t)(jt + crow) * CCH + cseg;
            unsigned dst = lds_base + (unsigned)(crow * BROW_STRIDE + cseg) * 2u;
#pragma unroll
            for (int q = 0; q < 4; ++q)
                async_copy_b128(dst + q * 16, src + q * 8);
        }
        wait_async0();
        __syncthreads();

        v8f acc[2][4];
        v8f zero = {};
#pragma unroll
        for (int mi = 0; mi < 2; ++mi)
#pragma unroll
            for (int ni = 0; ni < 4; ++ni) acc[mi][ni] = zero;

#pragma unroll 1
        for (int kc = 0; kc < NCHUNK; ++kc) {
            const int buf = kc & 1;
            // ---- async-prefetch next chunk into the other buffer ----
            if (kc + 1 < NCHUNK) {
                const unsigned short* src =
                    Bn + (size_t)(jt + crow) * CCH + (kc + 1) * KCHUNK + cseg;
                unsigned dst = lds_base +
                    (unsigned)(((buf ^ 1) * CHUNK_HALFS) + crow * BROW_STRIDE + cseg) * 2u;
#pragma unroll
                for (int q = 0; q < 4; ++q)
                    async_copy_b128(dst + q * 16, src + q * 8);
            }

            // ---- compute on current buffer: two 16x16x32 K-steps ----
            const unsigned short* bs = bsmem + buf * CHUNK_HALFS;
#pragma unroll
            for (int ks = 0; ks < 2; ++ks) {
                const int kglob = kc * KCHUNK + ks * 32;
                union Frag { v16bf v; int4 q[2]; };
                Frag ua[2], ub[4];
#pragma unroll
                for (int mi = 0; mi < 2; ++mi) {
                    ua[mi].q[0] = *(const int4*)(aptr[mi] + kglob + hsel * 8);
                    ua[mi].q[1] = *(const int4*)(aptr[mi] + kglob + 16 + hsel * 8);
                }
#pragma unroll
                for (int ni = 0; ni < 4; ++ni) {
                    const unsigned short* bp =
                        bs + (wn * 64 + ni * 16 + l15) * BROW_STRIDE + ks * 32 + hsel * 8;
                    ub[ni].q[0] = *(const int4*)(bp);
                    ub[ni].q[1] = *(const int4*)(bp + 16);
                }
#pragma unroll
                for (int mi = 0; mi < 2; ++mi)
#pragma unroll
                    for (int ni = 0; ni < 4; ++ni)
                        acc[mi][ni] = __builtin_amdgcn_wmma_f32_16x16x32_bf16(
                            false, ua[mi].v, false, ub[ni].v,
                            (short)0, acc[mi][ni], false, false);
            }

            // chunk kc+1 landed; all waves done reading buf before it is refilled
            wait_async0();
            __syncthreads();
        }

        // Fold this 128-wide j-tile into the running per-row (max, argmax).
#pragma unroll
        for (int mi = 0; mi < 2; ++mi)
#pragma unroll
            for (int ni = 0; ni < 4; ++ni) {
                const int j = jt + wn * 64 + ni * 16 + l15;
#pragma unroll
                for (int r = 0; r < 8; ++r) {
                    float v = acc[mi][ni][r];
                    if (v > rmax[mi][r]) { rmax[mi][r] = v; ridx[mi][r] = j; }
                }
            }
    }

    // Cross-lane / cross-wave reduction (aliases the B staging LDS).
    float* smax = (float*)smem;
    int*   sidx = (int*)(smem + 128 * 32 * 4);
#pragma unroll
    for (int mi = 0; mi < 2; ++mi)
#pragma unroll
        for (int r = 0; r < 8; ++r) {
            int row  = wm * 32 + mi * 16 + hsel * 8 + r;  // C-matrix: M = r + 8*hsel
            int slot = wn * 16 + l15;                     // C-matrix: N = l15
            smax[row * 32 + slot] = rmax[mi][r];
            sidx[row * 32 + slot] = ridx[mi][r];
        }
    __syncthreads();
    if (tid < 128) {
        float bm = smax[tid * 32];
        int   bi = sidx[tid * 32];
        for (int s = 1; s < 32; ++s) {
            float v = smax[tid * 32 + s];
            if (v > bm) { bm = v; bi = sidx[tid * 32 + s]; }
        }
        int i = blockIdx.x * 128 + tid;
        pmax[(size_t)blockIdx.y * HW + i] = bm;
        pidx[(size_t)blockIdx.y * HW + i] = bi;
    }
}

// ---------------------------------------------------------------------------
// 4) Final loss: merge partials, un-normalize max dot, mean(1 - cos).
// ---------------------------------------------------------------------------
__global__ void zero_out_kernel(float* out) { out[0] = 0.0f; }

__global__ void loss_kernel(const float* __restrict__ pmax, const int* __restrict__ pidx,
                            const float* __restrict__ normA, const float* __restrict__ denomA,
                            const float* __restrict__ normB, const float* __restrict__ denomB,
                            float* __restrict__ out) {
    __shared__ float red[256];
    int i = blockIdx.x * 256 + threadIdx.x;
    float bm = pmax[i];
    int   bi = pidx[i];
#pragma unroll
    for (int g = 1; g < 4; ++g) {
        float v = pmax[(size_t)g * HW + i];
        if (v > bm) { bm = v; bi = pidx[(size_t)g * HW + i]; }
    }
    // bm = dot(a_n[i], b_n[bi]); un-normalize to raw dot(a_i, b_bi)
    float dot = bm * denomA[i] * denomB[bi];
    float cs  = dot / ((normA[i] + EPSF) * (normB[bi] + EPSF));
    red[threadIdx.x] = 1.0f - cs;
    __syncthreads();
    for (int s = 128; s > 0; s >>= 1) {
        if (threadIdx.x < s) red[threadIdx.x] += red[threadIdx.x + s];
        __syncthreads();
    }
    if (threadIdx.x == 0) atomicAdd(out, red[0] * (1.0f / HW));
}

// ---------------------------------------------------------------------------
extern "C" void kernel_launch(void* const* d_in, const int* in_sizes, int n_in,
                              void* d_out, int out_size, void* d_ws, size_t ws_size,
                              hipStream_t stream) {
    const float* a = (const float*)d_in[0];
    const float* b = (const float*)d_in[1];
    float* out = (float*)d_out;

    char* ws = (char*)d_ws;
    const size_t bfBytes = (size_t)HW * CCH * sizeof(unsigned short); // 24 MB
    unsigned short* An = (unsigned short*)(ws);
    unsigned short* Bn = (unsigned short*)(ws + bfBytes);
    size_t off = 2 * bfBytes;
    float* normA  = (float*)(ws + off); off += (size_t)HW * 4;
    float* denomA = (float*)(ws + off); off += (size_t)HW * 4;
    float* normB  = (float*)(ws + off); off += (size_t)HW * 4;
    float* denomB = (float*)(ws + off); off += (size_t)HW * 4;
    float* pmax   = (float*)(ws + off); off += (size_t)4 * HW * 4;
    int*   pidx   = (int*)  (ws + off);

    // 1) norms
    norms_kernel<<<HW / 256, 256, 0, stream>>>(a, b, normA, denomA, normB, denomB);

    // 2) normalize + transpose to bf16 [pos][ch]
    dim3 xg(HW / 32, CCH / 32);
    xpose_kernel<<<xg, 256, 0, stream>>>(a, denomA, An);
    xpose_kernel<<<xg, 256, 0, stream>>>(b, denomB, Bn);

    // 3) WMMA argmax-GEMM: 128 i-tiles x 4 j-groups
    argmax_gemm_kernel<<<dim3(128, 4), 256, 0, stream>>>(An, Bn, pmax, pidx);

    // 4) loss
    zero_out_kernel<<<1, 1, 0, stream>>>(out);
    loss_kernel<<<HW / 256, 256, 0, stream>>>(pmax, pidx, normA, denomA, normB, denomB, out);
}
